// Encoder_10728828305502
// MI455X (gfx1250) — matrix operand order repaired and checked by reading, not compiled, not run on hardware
//
#include <hip/hip_runtime.h>
#include <math.h>

#define HDIM   512
#define NHEAD  8
#define DHEAD  64
#define NLAYER 6
#define SEQ    2048
#define NBATCH 2
#define MTOT   (NBATCH*SEQ)   // 4096
#define FFND   (4*HDIM)       // 2048

typedef __bf16 bf16_t;
typedef __attribute__((ext_vector_type(16))) __bf16 v16bf;
typedef __attribute__((ext_vector_type(8)))  __bf16 v8bf;
typedef __attribute__((ext_vector_type(8)))  float  v8f;

union BFPack { v16bf v; v8bf h[2]; };

// ---- WMMA fragment loaders (CDNA5 16-bit layouts, wave32) --------------------
// A (16x32, MxK): lane l: m=l&15, khi=l>>4; elems 0..7 -> K=8*khi..+7,
//                 elems 8..15 -> K=16+8*khi..+7  (two 16B chunks)
__device__ __forceinline__ v16bf load_a_frag(const bf16_t* base, int ld) {
  const int lane = threadIdx.x & 31;
  const int m = lane & 15, khi = lane >> 4;
  const bf16_t* p = base + (size_t)m * ld + 8 * khi;
  BFPack r;
  r.h[0] = *(const v8bf*)(p);
  r.h[1] = *(const v8bf*)(p + 16);
  return r.v;
}
// B (32x16, KxN): lane l: n=l&15, khi=l>>4; elems 0..7 -> K=8*khi..+7,
// elems 8..15 -> K=16+8*khi..+7. "base" must be row-major [n][k] (i.e. W^T).
__device__ __forceinline__ v16bf load_b_frag(const bf16_t* base, int ld) {
  const int lane = threadIdx.x & 31;
  const int n = lane & 15, khi = lane >> 4;
  const bf16_t* p = base + (size_t)n * ld + 8 * khi;
  BFPack r;
  r.h[0] = *(const v8bf*)(p);
  r.h[1] = *(const v8bf*)(p + 16);
  return r.v;
}

__device__ __forceinline__ v8f wmma_bf16(v16bf a, v16bf b, v8f c) {
  return __builtin_amdgcn_wmma_f32_16x16x32_bf16(false, a, false, b, (short)0, c,
                                                 false, false);
}
__device__ __forceinline__ v8f vzero8() {
  v8f z = {0.f,0.f,0.f,0.f,0.f,0.f,0.f,0.f};
  return z;
}

// ---- weight convert + transpose: W[K][N] f32 -> Wt[N][K] bf16 ----------------
__global__ __launch_bounds__(256)
void convert_w_kernel(const float* __restrict__ W, bf16_t* __restrict__ Wt,
                      int K, int N) {
  int idx = blockIdx.x * 256 + threadIdx.x;
  if (idx >= K * N) return;
  int k = idx / N, n = idx % N;
  Wt[(size_t)n * K + k] = (bf16_t)W[(size_t)k * N + n];
}

// ---- embedding + sinusoidal positional encoding ------------------------------
__global__ __launch_bounds__(256)
void embed_kernel(const int* __restrict__ src, const float* __restrict__ emb,
                  float* __restrict__ x, bf16_t* __restrict__ xb) {
  int row = blockIdx.x, tid = threadIdx.x;      // row = b*SEQ + s
  int s = row & (SEQ - 1);
  int tok = src[row];
  // pair index p = tid; div = 10000^(2p/H); even col -> sin, odd col -> cos
  float div = powf(10000.0f, (float)tid / 256.0f);
  float ang = (float)s / div;
  float e0 = emb[(size_t)tok * HDIM + 2 * tid]     + sinf(ang);
  float e1 = emb[(size_t)tok * HDIM + 2 * tid + 1] + cosf(ang);
  size_t o = (size_t)row * HDIM + 2 * tid;
  x[o] = e0; x[o + 1] = e1;
  xb[o] = (bf16_t)e0; xb[o + 1] = (bf16_t)e1;
}

// ---- GEMM: C[M,N] = A[M,K](bf16) * Wt[N,K]^T(bf16) + bias --------------------
// one wave = 64x64 tile; epilogue modes:
//  0: f32 row-major   1: bf16 row-major (+relu)   2: bf16 [B,NH,S,DH]   3: bf16 [B,NH,DH,S]
__global__ __launch_bounds__(256)
void gemm_wmma(const bf16_t* __restrict__ A, const bf16_t* __restrict__ Wt,
               const float* __restrict__ bias, void* __restrict__ out,
               int M, int N, int K, int mode, int relu) {
  int wave = (blockIdx.x * 256 + threadIdx.x) >> 5;
  int tiles_n = N >> 6;
  int tm = wave / tiles_n, tn = wave - tm * tiles_n;
  int m0 = tm * 64, n0 = tn * 64;
  if (m0 >= M) return;

  const int lane = threadIdx.x & 31;
  const int nloc = lane & 15, khi = lane >> 4;

  v8f acc[4][4];
#pragma unroll
  for (int mi = 0; mi < 4; mi++)
#pragma unroll
    for (int ni = 0; ni < 4; ni++) acc[mi][ni] = vzero8();

  for (int k0 = 0; k0 < K; k0 += 32) {
    v16bf af[4], bfv[4];
#pragma unroll
    for (int mi = 0; mi < 4; mi++)
      af[mi] = load_a_frag(A + (size_t)(m0 + 16 * mi) * K + k0, K);
#pragma unroll
    for (int ni = 0; ni < 4; ni++)
      bfv[ni] = load_b_frag(Wt + (size_t)(n0 + 16 * ni) * K + k0, K);
#pragma unroll
    for (int mi = 0; mi < 4; mi++)
#pragma unroll
      for (int ni = 0; ni < 4; ni++)
        acc[mi][ni] = wmma_bf16(af[mi], bfv[ni], acc[mi][ni]);
  }

#pragma unroll
  for (int mi = 0; mi < 4; mi++) {
#pragma unroll
    for (int ni = 0; ni < 4; ni++) {
      int col = n0 + 16 * ni + nloc;
      float bv = bias ? bias[col] : 0.0f;
#pragma unroll
      for (int r = 0; r < 8; r++) {
        int row = m0 + 16 * mi + r + 8 * khi;
        float v = acc[mi][ni][r] + bv;
        if (relu) v = fmaxf(v, 0.0f);
        if (mode == 0) {
          ((float*)out)[(size_t)row * N + col] = v;
        } else if (mode == 1) {
          ((bf16_t*)out)[(size_t)row * N + col] = (bf16_t)v;
        } else {
          int b = row >> 11, s = row & (SEQ - 1);
          int h = col >> 6, d = col & (DHEAD - 1);
          size_t idx = (mode == 2)
            ? (((size_t)(b * NHEAD + h) * SEQ + s) * DHEAD + d)     // Q/K: [b,h,s,d]
            : (((size_t)(b * NHEAD + h) * DHEAD + d) * SEQ + s);    // V^T: [b,h,d,s]
          ((bf16_t*)out)[idx] = (bf16_t)v;
        }
      }
    }
  }
}

// ---- flash attention: one wave per (b,h,16 query rows) -----------------------
__global__ __launch_bounds__(32)
void attn_kernel(const bf16_t* __restrict__ Q, const bf16_t* __restrict__ Km,
                 const bf16_t* __restrict__ Vt, const int* __restrict__ mask,
                 bf16_t* __restrict__ AV) {
  __shared__ __align__(16) bf16_t sP[16 * 32];
  const int lane = threadIdx.x;
  const int nloc = lane & 15, khi = lane >> 4;
  int qt = blockIdx.x & (SEQ / 16 - 1);
  int bh = blockIdx.x / (SEQ / 16);
  int h = bh & (NHEAD - 1), b = bh / NHEAD;
  int q0 = qt * 16;

  const bf16_t* Qb = Q + ((size_t)bh * SEQ + q0) * DHEAD;
  v16bf qf0 = load_a_frag(Qb, DHEAD);
  v16bf qf1 = load_a_frag(Qb + 32, DHEAD);

  float rmax[8], rsum[8];
  v8f accd[4];
#pragma unroll
  for (int r = 0; r < 8; r++) { rmax[r] = -3.0e38f; rsum[r] = 0.0f; }
#pragma unroll
  for (int t = 0; t < 4; t++) accd[t] = vzero8();

  for (int s0 = 0; s0 < SEQ; s0 += 32) {
    const bf16_t* Kb = Km + ((size_t)bh * SEQ + s0) * DHEAD;
    v8f slo = vzero8(), shi = vzero8();
    slo = wmma_bf16(qf0, load_b_frag(Kb, DHEAD), slo);
    slo = wmma_bf16(qf1, load_b_frag(Kb + 32, DHEAD), slo);
    shi = wmma_bf16(qf0, load_b_frag(Kb + 16 * DHEAD, DHEAD), shi);
    shi = wmma_bf16(qf1, load_b_frag(Kb + 16 * DHEAD + 32, DHEAD), shi);

    int mlo = mask[b * SEQ + s0 + nloc];
    int mhi = mask[b * SEQ + s0 + 16 + nloc];

#pragma unroll
    for (int r = 0; r < 8; r++) {
      float lo = slo[r] * 0.125f; if (mlo == 0) lo = -1.0e20f;
      float hi = shi[r] * 0.125f; if (mhi == 0) hi = -1.0e20f;
      float mx = fmaxf(lo, hi);
#pragma unroll
      for (int off = 1; off < 16; off <<= 1)
        mx = fmaxf(mx, __shfl_xor(mx, off, 32));
      float nm = fmaxf(rmax[r], mx);
      float corr = __expf(rmax[r] - nm);
      rmax[r] = nm;
      float plo = __expf(lo - nm), phi = __expf(hi - nm);
      float ps = plo + phi;
#pragma unroll
      for (int off = 1; off < 16; off <<= 1)
        ps += __shfl_xor(ps, off, 32);
      rsum[r] = rsum[r] * corr + ps;
#pragma unroll
      for (int t = 0; t < 4; t++) accd[t][r] *= corr;
      int m = r + 8 * khi;
      sP[m * 32 + nloc]      = (bf16_t)plo;
      sP[m * 32 + 16 + nloc] = (bf16_t)phi;
    }
    __syncthreads();
    v16bf pf = load_a_frag(sP, 32);
    const bf16_t* Vb = Vt + (size_t)bh * DHEAD * SEQ + s0;
#pragma unroll
    for (int t = 0; t < 4; t++)
      accd[t] = wmma_bf16(pf, load_b_frag(Vb + (size_t)(16 * t) * SEQ, SEQ), accd[t]);
    __syncthreads();
  }

#pragma unroll
  for (int t = 0; t < 4; t++) {
#pragma unroll
    for (int r = 0; r < 8; r++) {
      int m = r + 8 * khi;
      float v = accd[t][r] / rsum[r];
      size_t idx = ((size_t)(b * SEQ + q0 + m)) * HDIM + h * DHEAD + 16 * t + nloc;
      AV[idx] = (bf16_t)v;
    }
  }
}

// ---- residual + LayerNorm: out = resin + LN(y)*g + b -------------------------
__global__ __launch_bounds__(256)
void ln_residual_kernel(const float* __restrict__ resin, const float* __restrict__ y,
                        const float* __restrict__ gamma, const float* __restrict__ beta,
                        float* __restrict__ xout, bf16_t* __restrict__ xbout) {
  __shared__ float s1[256], s2[256];
  int row = blockIdx.x, tid = threadIdx.x;
  size_t o = (size_t)row * HDIM + 2 * tid;
  float a = y[o], c = y[o + 1];
  s1[tid] = a + c; s2[tid] = a * a + c * c;
  __syncthreads();
  for (int st = 128; st > 0; st >>= 1) {
    if (tid < st) { s1[tid] += s1[tid + st]; s2[tid] += s2[tid + st]; }
    __syncthreads();
  }
  float mu = s1[0] * (1.0f / HDIM);
  float var = s2[0] * (1.0f / HDIM) - mu * mu;
  float rs = rsqrtf(var + 1e-5f);
  float r0 = resin[o]     + (a - mu) * rs * gamma[2 * tid]     + beta[2 * tid];
  float r1 = resin[o + 1] + (c - mu) * rs * gamma[2 * tid + 1] + beta[2 * tid + 1];
  xout[o] = r0; xout[o + 1] = r1;
  xbout[o] = (bf16_t)r0; xbout[o + 1] = (bf16_t)r1;
}

// -----------------------------------------------------------------------------
static inline void launch_gemm(const bf16_t* A, const bf16_t* Wt, const float* bias,
                               void* out, int M, int N, int K, int mode, int relu,
                               hipStream_t s) {
  int blocks = (M / 64) * (N / 64) / 8;   // 8 waves per block
  gemm_wmma<<<blocks, 256, 0, s>>>(A, Wt, bias, out, M, N, K, mode, relu);
}

extern "C" void kernel_launch(void* const* d_in, const int* in_sizes, int n_in,
                              void* d_out, int out_size, void* d_ws, size_t ws_size,
                              hipStream_t stream) {
  (void)in_sizes; (void)n_in; (void)out_size; (void)ws_size;
  const int*   src  = (const int*)d_in[0];
  const int*   mask = (const int*)d_in[1];
  const float* emb  = (const float*)d_in[2];
  const float* Wq   = (const float*)d_in[3];
  const float* bq   = (const float*)d_in[4];
  const float* Wk   = (const float*)d_in[5];
  const float* bk   = (const float*)d_in[6];
  const float* Wv   = (const float*)d_in[7];
  const float* bv   = (const float*)d_in[8];
  const float* Wo   = (const float*)d_in[9];
  const float* bo   = (const float*)d_in[10];
  const float* gamma= (const float*)d_in[11];
  const float* beta = (const float*)d_in[12];
  const float* W1   = (const float*)d_in[13];
  const float* b1   = (const float*)d_in[14];
  const float* W2   = (const float*)d_in[15];
  const float* b2   = (const float*)d_in[16];

  // --- carve workspace -------------------------------------------------------
  char* p = (char*)d_ws;
  auto alloc = [&](size_t bytes) -> char* {
    char* r = p; p += (bytes + 255) & ~(size_t)255; return r;
  };
  const size_t HH = (size_t)HDIM * HDIM, HF = (size_t)HDIM * FFND;
  bf16_t* WqT = (bf16_t*)alloc(NLAYER * HH * 2);
  bf16_t* WkT = (bf16_t*)alloc(NLAYER * HH * 2);
  bf16_t* WvT = (bf16_t*)alloc(NLAYER * HH * 2);
  bf16_t* WoT = (bf16_t*)alloc(NLAYER * HH * 2);
  bf16_t* W1T = (bf16_t*)alloc(NLAYER * HF * 2);
  bf16_t* W2T = (bf16_t*)alloc(NLAYER * HF * 2);
  float*  xbuf   = (float*)alloc((size_t)MTOT * HDIM * 4);
  float*  resbuf = (float*)alloc((size_t)MTOT * HDIM * 4);
  float*  ybuf   = (float*)alloc((size_t)MTOT * HDIM * 4);
  bf16_t* xb     = (bf16_t*)alloc((size_t)MTOT * HDIM * 2);
  bf16_t* resb   = (bf16_t*)alloc((size_t)MTOT * HDIM * 2);
  bf16_t* qbuf   = (bf16_t*)alloc((size_t)MTOT * HDIM * 2);
  bf16_t* kbuf   = (bf16_t*)alloc((size_t)MTOT * HDIM * 2);
  bf16_t* vtbuf  = (bf16_t*)alloc((size_t)MTOT * HDIM * 2);
  bf16_t* avbuf  = (bf16_t*)alloc((size_t)MTOT * HDIM * 2);
  bf16_t* h1buf  = (bf16_t*)alloc((size_t)MTOT * FFND * 2);

  // --- weights -> bf16, transposed to [N][K] ---------------------------------
  for (int i = 0; i < NLAYER; i++) {
    int bHH = (int)((HH + 255) / 256), bHF = (int)((HF + 255) / 256);
    convert_w_kernel<<<bHH, 256, 0, stream>>>(Wq + i * HH, WqT + i * HH, HDIM, HDIM);
    convert_w_kernel<<<bHH, 256, 0, stream>>>(Wk + i * HH, WkT + i * HH, HDIM, HDIM);
    convert_w_kernel<<<bHH, 256, 0, stream>>>(Wv + i * HH, WvT + i * HH, HDIM, HDIM);
    convert_w_kernel<<<bHH, 256, 0, stream>>>(Wo + i * HH, WoT + i * HH, HDIM, HDIM);
    convert_w_kernel<<<bHF, 256, 0, stream>>>(W1 + i * HF, W1T + i * HF, HDIM, FFND);
    convert_w_kernel<<<bHF, 256, 0, stream>>>(W2 + i * HF, W2T + i * HF, FFND, HDIM);
  }

  embed_kernel<<<MTOT, 256, 0, stream>>>(src, emb, xbuf, xb);

  for (int i = 0; i < NLAYER; i++) {
    launch_gemm(xb, WqT + i * HH, bq + i * HDIM, qbuf,  MTOT, HDIM, HDIM, 2, 0, stream);
    launch_gemm(xb, WkT + i * HH, bk + i * HDIM, kbuf,  MTOT, HDIM, HDIM, 2, 0, stream);
    launch_gemm(xb, WvT + i * HH, bv + i * HDIM, vtbuf, MTOT, HDIM, HDIM, 3, 0, stream);

    attn_kernel<<<NBATCH * NHEAD * (SEQ / 16), 32, 0, stream>>>(qbuf, kbuf, vtbuf,
                                                                mask, avbuf);

    launch_gemm(avbuf, WoT + i * HH, bo + i * HDIM, ybuf, MTOT, HDIM, HDIM, 0, 0, stream);
    ln_residual_kernel<<<MTOT, 256, 0, stream>>>(xbuf, ybuf, gamma + i * HDIM,
                                                 beta + i * HDIM, resbuf, resb);

    launch_gemm(resb,  W1T + i * HF, b1 + (size_t)i * FFND, h1buf, MTOT, FFND, HDIM, 1, 1, stream);
    launch_gemm(h1buf, W2T + i * HF, b2 + i * HDIM,          ybuf, MTOT, HDIM, FFND, 0, 0, stream);

    float* xdst = (i == NLAYER - 1) ? (float*)d_out : xbuf;
    ln_residual_kernel<<<MTOT, 256, 0, stream>>>(resbuf, ybuf, gamma + i * HDIM,
                                                 beta + i * HDIM, xdst, xb);
  }
}
